// LabelSequence_9320079032646
// MI455X (gfx1250) — matrix operand order repaired and checked by reading, not compiled, not run on hardware
//
#include <hip/hip_runtime.h>
#include <hip/hip_bf16.h>

// ---------------- problem constants ----------------
constexpr int Bn = 64;
constexpr int Tn = 512;
constexpr int En = 256;
constexpr int Fn = 256;
constexpr int Hn = 256;
constexpr int Ln = 9;
constexpr int Lp = 16;                 // L padded to one WMMA N-tile
constexpr int BT = Bn * Tn;            // 32768
constexpr float KEEP = 0.9f;

// per-wave macro-tile: 32 rows (2 M-tiles) x 64 cols (4 N-tiles)
constexpr int MT = 2;
constexpr int NT = 4;

// ---------------- WMMA types (CDNA5, wave32) ----------------
typedef __attribute__((ext_vector_type(16))) __bf16 v16bf;
typedef __attribute__((ext_vector_type(8)))  float  v8f;

union AFrag { v16bf v; uint4 q[2]; unsigned short u[16]; };
union BFrag { v16bf v; uint4 q[2]; unsigned short u[16]; };
union CFrag { v8f   v; float f[8]; };

__device__ __forceinline__ v8f wmma_bf16(v8f c, v16bf a, v16bf b) {
    // (neg_a, A, neg_b, B, c_mod, C, reuse_a, reuse_b)
    return __builtin_amdgcn_wmma_f32_16x16x32_bf16(false, a, false, b, (short)0, c, false, false);
}

__device__ __forceinline__ unsigned short f2bf(float f) {
    unsigned u = __float_as_uint(f);
    unsigned r = (u + 0x7FFFu + ((u >> 16) & 1u)) >> 16;   // RNE
    return (unsigned short)r;
}

__device__ __forceinline__ unsigned hash32(unsigned x) {
    x ^= x >> 16; x *= 0x7feb352du;
    x ^= x >> 15; x *= 0x846ca68bu;
    x ^= x >> 16; return x;
}

// LDS byte offset of a __shared__ object (generic -> addrspace(3) -> int)
__device__ __forceinline__ unsigned lds_off(const void* p) {
    return (unsigned)(unsigned long long)(__attribute__((address_space(3))) const void*)p;
}

// ---------------- weight prep: f32 -> bf16, B-operands transposed to (N,K) ----------------
__global__ void prep_weights(const float* __restrict__ Wl, const float* __restrict__ Wh,
                             const float* __restrict__ w0, const float* __restrict__ w1,
                             const float* __restrict__ w2, const float* __restrict__ Wfc,
                             unsigned short* __restrict__ wlT,   // [Fn][En]  wlT[n*En+k] = Wl[k*Fn+n]
                             unsigned short* __restrict__ whT,   // [Hn][Fn]
                             unsigned short* __restrict__ wtap,  // [3 conv][3 tap][Fout][Fin]
                             unsigned short* __restrict__ wfcT)  // [Lp][Hn], rows >= Ln zero
{
    int tid = blockIdx.x * blockDim.x + threadIdx.x;
    int stride = gridDim.x * blockDim.x;
    for (int idx = tid; idx < Fn * En; idx += stride) {
        int n = idx / En, k = idx % En;
        wlT[idx] = f2bf(Wl[k * Fn + n]);
    }
    for (int idx = tid; idx < Hn * Fn; idx += stride) {
        int n = idx / Fn, k = idx % Fn;
        whT[idx] = f2bf(Wh[k * Hn + n]);
    }
    const float* ws[3] = { w0, w1, w2 };
    for (int idx = tid; idx < 9 * Fn * Fn; idx += stride) {
        int slot = idx / (Fn * Fn);     // conv*3 + tap
        int c = slot / 3, tap = slot % 3;
        int e = idx % (Fn * Fn);
        int o = e / Fn, ii = e % Fn;
        wtap[idx] = f2bf(ws[c][(o * Fn + ii) * 3 + tap]);   // w[o, i, tap] (OIH)
    }
    for (int idx = tid; idx < Lp * Hn; idx += stride) {
        int n = idx / Hn, k = idx % Hn;
        wfcT[idx] = (n < Ln) ? f2bf(Wfc[k * Ln + n]) : (unsigned short)0;
    }
}

// ---------------- embedding gather + f32->bf16 (each element converted exactly once) --------
__global__ void gather_embed(const int* __restrict__ ids, const float* __restrict__ Wemb,
                             unsigned short* __restrict__ xE)
{
    int idx = blockIdx.x * blockDim.x + threadIdx.x;   // BT * (En/4) threads
    int g  = idx >> 6;                                  // token row
    int c4 = (idx & 63) << 2;                           // column group of 4
    int tok = ids[g];
    float4 f = *(const float4*)(Wemb + (long long)tok * En + c4);
    ushort4 o;
    o.x = f2bf(f.x); o.y = f2bf(f.y); o.z = f2bf(f.z); o.w = f2bf(f.w);
    *(ushort4*)(xE + (long long)g * En + c4) = o;
}

// ---------------- generic dense GEMM: (BT,256)bf16 @ Bt(N,K)bf16 -> 32x64 per wave ----------
// MODE 0: + bias, store bf16.
// MODE 1: + bias, fold 4 dropout rounds into effective mask, store bf16.
template<int MODE>
__global__ __launch_bounds__(256) void linear_gemm(const unsigned short* __restrict__ A,
                                                   const unsigned short* __restrict__ Bt,
                                                   const float* __restrict__ bias,
                                                   unsigned short* __restrict__ outp)
{
    int wave = (blockIdx.x * blockDim.x + threadIdx.x) >> 5;
    int lane = threadIdx.x & 31;
    int mg = wave >> 2;                 // 0..1023  (32-row group)
    int ng = wave & 3;                  // 0..3     (64-col group)
    int r  = lane & 15;
    int hi = lane >> 4;
    int m0 = mg * 32;
    int n0 = ng * 64;

    const unsigned short* aptr[MT];
    #pragma unroll
    for (int mi = 0; mi < MT; ++mi)
        aptr[mi] = A + (long long)(m0 + mi * 16 + r) * Fn + hi * 16;
    const unsigned short* bptr[NT];
    #pragma unroll
    for (int j = 0; j < NT; ++j)
        bptr[j] = Bt + (long long)(n0 + j * 16 + r) * Fn + hi * 16;

    CFrag acc[MT][NT];
    #pragma unroll
    for (int mi = 0; mi < MT; ++mi)
        #pragma unroll
        for (int j = 0; j < NT; ++j)
            #pragma unroll
            for (int i = 0; i < 8; ++i) acc[mi][j].f[i] = 0.0f;

    for (int k0 = 0; k0 < Fn; k0 += 32) {
        AFrag a[MT];
        #pragma unroll
        for (int mi = 0; mi < MT; ++mi) {
            a[mi].q[0] = *(const uint4*)(aptr[mi] + k0);
            a[mi].q[1] = *(const uint4*)(aptr[mi] + k0 + 8);
        }
        BFrag b[NT];
        #pragma unroll
        for (int j = 0; j < NT; ++j) {
            b[j].q[0] = *(const uint4*)(bptr[j] + k0);
            b[j].q[1] = *(const uint4*)(bptr[j] + k0 + 8);
        }
        #pragma unroll
        for (int mi = 0; mi < MT; ++mi)
            #pragma unroll
            for (int j = 0; j < NT; ++j)
                acc[mi][j].v = wmma_bf16(acc[mi][j].v, a[mi].v, b[j].v);
    }

    #pragma unroll
    for (int j = 0; j < NT; ++j) {
        int nc = n0 + j * 16 + r;
        float bv = bias[nc];
        #pragma unroll
        for (int mi = 0; mi < MT; ++mi) {
            #pragma unroll
            for (int i = 0; i < 8; ++i) {
                long long row = m0 + mi * 16 + hi * 8 + i;
                float v = acc[mi][j].f[i] + bv;
                if (MODE == 1) {
                    // one hash -> 4 independent byte-tests = 4 dropout rounds
                    unsigned h = hash32((unsigned)(row * Hn + nc) * 0x9E3779B9u + 0x85ebca6bu);
                    int cnt = ((h & 0xffu) < 230u) + (((h >> 8) & 0xffu) < 230u) +
                              (((h >> 16) & 0xffu) < 230u) + ((h >> 24) < 230u);
                    v *= (float)cnt * (1.0f / (4.0f * KEEP));
                }
                outp[row * Fn + nc] = f2bf(v);
            }
        }
    }
}

// ---------------- conv1d (kernel 3, dilation d): async-stage A panel to LDS, WMMA from LDS --
// Block = 64 activation rows (2 waves' worth of M) x all 256 output cols (4 n-groups).
// A panel: rows [t0-2, t0+65] (zero halo) staged once via global_load_async_to_lds_b128,
// then reused by all 8 waves, all 3 taps, all k-steps from LDS.
constexpr int SROW = 264;              // row stride in ushorts (+16B pad: conflict-free b128)
constexpr int HALO = 2;                // max dilation
__global__ __launch_bounds__(256) void conv_gemm(const unsigned short* __restrict__ X,
                                                 unsigned short* __restrict__ Y,
                                                 const unsigned short* __restrict__ Wt, // [3][Fout][Fin]
                                                 const float* __restrict__ bias,
                                                 int dil, int do_relu)
{
    __shared__ __align__(16) unsigned short sA[(64 + 2 * HALO) * SROW];

    int tid = threadIdx.x;
    int m0B = blockIdx.x * 64;
    int t0B = m0B % Tn;                 // 64-row block never crosses a batch (512 % 64 == 0)
    int bb  = m0B - t0B;

    // ---- stage A rows [t0B-HALO, t0B+63+HALO] into LDS; zeros outside [0,Tn) ----
    for (int idx = tid; idx < (64 + 2 * HALO) * (Fn / 8); idx += 256) {
        int row = idx >> 5;                       // 0..67
        int c   = (idx & 31) * 8;                 // ushort column, 16B chunks
        int t   = t0B + row - HALO;
        unsigned short* dstp = &sA[row * SROW + c];
        if (t >= 0 && t < Tn) {
            const unsigned short* src = X + (long long)(bb + t) * Fn + c;
            asm volatile("global_load_async_to_lds_b128 %0, %1, off"
                         :: "v"(lds_off(dstp)),
                            "v"((unsigned long long)(uintptr_t)src)
                         : "memory");
        } else {
            *(uint4*)dstp = make_uint4(0u, 0u, 0u, 0u);
        }
    }
    asm volatile("s_wait_asynccnt 0" ::: "memory");
    __syncthreads();

    // ---- per-wave 32x64 WMMA tile, A from LDS, B from global (L2-hot weights) ----
    int w    = tid >> 5;
    int lane = tid & 31;
    int mig  = w >> 2;                  // 0..1
    int ng   = w & 3;                   // 0..3
    int r    = lane & 15;
    int hi   = lane >> 4;
    int m0   = m0B + mig * 32;
    int n0   = ng * 64;
    int srow0 = mig * 32 + r + HALO;    // this lane's A row in LDS (center tap)

    const unsigned short* bptr[NT];
    #pragma unroll
    for (int j = 0; j < NT; ++j)
        bptr[j] = Wt + (long long)(n0 + j * 16 + r) * Fn + hi * 16;

    CFrag acc[MT][NT];
    #pragma unroll
    for (int mi = 0; mi < MT; ++mi)
        #pragma unroll
        for (int j = 0; j < NT; ++j)
            #pragma unroll
            for (int i = 0; i < 8; ++i) acc[mi][j].f[i] = 0.0f;

    for (int tap = 0; tap < 3; ++tap) {
        int off = (tap - 1) * dil;
        const unsigned short* arow[MT];
        #pragma unroll
        for (int mi = 0; mi < MT; ++mi)
            arow[mi] = &sA[(srow0 + mi * 16 + off) * SROW + hi * 16];
        for (int k0 = 0; k0 < Fn; k0 += 32) {
            AFrag a[MT];
            #pragma unroll
            for (int mi = 0; mi < MT; ++mi) {
                a[mi].q[0] = *(const uint4*)(arow[mi] + k0);
                a[mi].q[1] = *(const uint4*)(arow[mi] + k0 + 8);
            }
            BFrag b[NT];
            #pragma unroll
            for (int j = 0; j < NT; ++j) {
                const unsigned short* bp = bptr[j] + (long long)tap * Fn * Fn;
                b[j].q[0] = *(const uint4*)(bp + k0);
                b[j].q[1] = *(const uint4*)(bp + k0 + 8);
            }
            #pragma unroll
            for (int mi = 0; mi < MT; ++mi)
                #pragma unroll
                for (int j = 0; j < NT; ++j)
                    acc[mi][j].v = wmma_bf16(acc[mi][j].v, a[mi].v, b[j].v);
        }
    }

    #pragma unroll
    for (int j = 0; j < NT; ++j) {
        int nc = n0 + j * 16 + r;
        float bv = bias[nc];
        #pragma unroll
        for (int mi = 0; mi < MT; ++mi) {
            #pragma unroll
            for (int i = 0; i < 8; ++i) {
                long long row = m0 + mi * 16 + hi * 8 + i;
                float v = acc[mi][j].f[i] + bv;
                if (do_relu) v = fmaxf(v, 0.0f);
                Y[row * Fn + nc] = f2bf(v);
            }
        }
    }
}

// ---------------- logits = outm @ WfcT(padded 16) + bfc, WMMA, 32x16 per wave ---------------
__global__ __launch_bounds__(256) void fc_wmma(const unsigned short* __restrict__ A,
                                               const unsigned short* __restrict__ wfcT,
                                               const float* __restrict__ bfc,
                                               float* __restrict__ logits)
{
    int wave = (blockIdx.x * blockDim.x + threadIdx.x) >> 5;  // 0..1023
    int lane = threadIdx.x & 31;
    int r  = lane & 15;
    int hi = lane >> 4;
    int m0 = wave * 32;

    const unsigned short* aptr[MT];
    #pragma unroll
    for (int mi = 0; mi < MT; ++mi)
        aptr[mi] = A + (long long)(m0 + mi * 16 + r) * Hn + hi * 16;
    const unsigned short* bptr = wfcT + (long long)r * Hn + hi * 16;

    CFrag acc[MT];
    #pragma unroll
    for (int mi = 0; mi < MT; ++mi)
        #pragma unroll
        for (int i = 0; i < 8; ++i) acc[mi].f[i] = 0.0f;

    for (int k0 = 0; k0 < Hn; k0 += 32) {
        AFrag a[MT];
        #pragma unroll
        for (int mi = 0; mi < MT; ++mi) {
            a[mi].q[0] = *(const uint4*)(aptr[mi] + k0);
            a[mi].q[1] = *(const uint4*)(aptr[mi] + k0 + 8);
        }
        BFrag b;
        b.q[0] = *(const uint4*)(bptr + k0);
        b.q[1] = *(const uint4*)(bptr + k0 + 8);
        #pragma unroll
        for (int mi = 0; mi < MT; ++mi)
            acc[mi].v = wmma_bf16(acc[mi].v, a[mi].v, b.v);
    }

    if (r < Ln) {
        float bv = bfc[r];
        #pragma unroll
        for (int mi = 0; mi < MT; ++mi)
            #pragma unroll
            for (int i = 0; i < 8; ++i) {
                long long row = m0 + mi * 16 + hi * 8 + i;
                logits[row * Ln + r] = acc[mi].f[i] + bv;
            }
    }
}

// ---------------- CRF: numerator + forward logsumexp scan; one wave per batch row ----------
__global__ void crf_loss(const int* __restrict__ ids_all, const int* __restrict__ lab_all,
                         const float* __restrict__ logits, const float* __restrict__ cstart,
                         const float* __restrict__ cend, const float* __restrict__ ctrans,
                         float* __restrict__ out)
{
    int b = blockIdx.x;
    int lane = threadIdx.x;
    const int* ids = ids_all + b * Tn;
    const int* lab = lab_all + b * Tn;
    const float* lg = logits + (long long)b * Tn * Ln;

    // sequence end = popcount(mask) - 1
    int cnt = 0;
    for (int t = lane; t < Tn; t += 32) cnt += (ids[t] > 0) ? 1 : 0;
    for (int o = 16; o > 0; o >>= 1) cnt += __shfl_down(cnt, o);
    cnt = __shfl(cnt, 0);
    int send = cnt - 1;

    // numerator (lane-parallel over t)
    float np = 0.0f;
    for (int t = 1 + lane; t < Tn; t += 32)
        if (ids[t] > 0)
            np += ctrans[lab[t - 1] * Ln + lab[t]] + lg[t * Ln + lab[t]];
    for (int o = 16; o > 0; o >>= 1) np += __shfl_down(np, o);
    float numer = 0.0f;
    if (lane == 0)
        numer = cstart[lab[0]] + lg[lab[0]] + np + cend[lab[send]];

    // forward scan: lanes 0..8 carry alpha_j
    float tc[Ln];
    float alpha = -1e30f;
    if (lane < Ln) {
        alpha = cstart[lane] + lg[lane];
        #pragma unroll
        for (int i = 0; i < Ln; ++i) tc[i] = ctrans[i * Ln + lane];
    }
    for (int t = 1; t < Tn; ++t) {
        if (ids[t] <= 0) continue;          // uniform per wave
        float em = (lane < Ln) ? lg[t * Ln + lane] : 0.0f;
        float v[Ln];
        float mx = -1e30f;
        #pragma unroll
        for (int i = 0; i < Ln; ++i) {
            float ai = __shfl(alpha, i);
            v[i] = ai + tc[i];
            mx = fmaxf(mx, v[i]);
        }
        float s = 0.0f;
        #pragma unroll
        for (int i = 0; i < Ln; ++i) s += __expf(v[i] - mx);
        float nxt = mx + __logf(s) + em;
        if (lane < Ln) alpha = nxt;
    }

    // denominator = logsumexp_j(alpha_j + end_j)
    float term = (lane < Ln) ? (alpha + cend[lane]) : -1e30f;
    float mx = term;
    for (int o = 16; o > 0; o >>= 1) mx = fmaxf(mx, __shfl_down(mx, o));
    mx = __shfl(mx, 0);
    float se = (lane < Ln) ? __expf(term - mx) : 0.0f;
    for (int o = 16; o > 0; o >>= 1) se += __shfl_down(se, o);
    if (lane == 0) {
        float denom = mx + __logf(se);
        atomicAdd(out, denom - numer);
    }
}

// ---------------- host orchestration ----------------
extern "C" void kernel_launch(void* const* d_in, const int* in_sizes, int n_in,
                              void* d_out, int out_size, void* d_ws, size_t ws_size,
                              hipStream_t stream)
{
    const int*   input_ids = (const int*)d_in[0];
    const int*   labels    = (const int*)d_in[1];
    const float* W_emb     = (const float*)d_in[2];
    const float* w0        = (const float*)d_in[3];
    const float* b0        = (const float*)d_in[4];
    const float* w1        = (const float*)d_in[5];
    const float* b1        = (const float*)d_in[6];
    const float* w2        = (const float*)d_in[7];
    const float* b2        = (const float*)d_in[8];
    const float* Wl        = (const float*)d_in[9];
    const float* bl        = (const float*)d_in[10];
    const float* Wh        = (const float*)d_in[11];
    const float* bh        = (const float*)d_in[12];
    const float* Wfc       = (const float*)d_in[13];
    const float* bfc       = (const float*)d_in[14];
    const float* crf_start = (const float*)d_in[15];
    const float* crf_end   = (const float*)d_in[16];
    const float* crf_trans = (const float*)d_in[17];

    // workspace carve-up
    size_t off = 0;
    auto carve = [&](size_t bytes) -> void* {
        void* p = (char*)d_ws + off;
        off += (bytes + 255) & ~(size_t)255;
        return p;
    };
    unsigned short* xE     = (unsigned short*)carve((size_t)BT * En * 2);
    unsigned short* xA     = (unsigned short*)carve((size_t)BT * Fn * 2);
    unsigned short* xB     = (unsigned short*)carve((size_t)BT * Fn * 2);
    unsigned short* outm   = (unsigned short*)carve((size_t)BT * Hn * 2);
    float*          logits = (float*)carve((size_t)BT * Ln * 4);
    unsigned short* wlT    = (unsigned short*)carve((size_t)Fn * En * 2);
    unsigned short* whT    = (unsigned short*)carve((size_t)Hn * Fn * 2);
    unsigned short* wtap   = (unsigned short*)carve((size_t)9 * Fn * Fn * 2);
    unsigned short* wfcT   = (unsigned short*)carve((size_t)Lp * Hn * 2);

    hipMemsetAsync(d_out, 0, sizeof(float), stream);

    prep_weights<<<256, 256, 0, stream>>>(Wl, Wh, w0, w1, w2, Wfc, wlT, whT, wtap, wfcT);

    gather_embed<<<(BT * (En / 4)) / 256, 256, 0, stream>>>(input_ids, W_emb, xE);

    // 1024 m-groups (32 rows) * 4 n-groups (64 cols) = 4096 waves; 8 waves / block
    const int GEMM_BLOCKS = (BT / 32) * (Fn / 64) / 8;   // 512
    linear_gemm<0><<<GEMM_BLOCKS, 256, 0, stream>>>(xE, wlT, bl, xA);

    // conv schedule: {weight idx, dilation, relu}
    struct Cfg { int c, dil, relu; };
    const Cfg seq[12] = {
        {0,1,1},{1,1,0},{2,2,1},   // block 1 (+outer relu fused into conv2)
        {0,1,1},{1,1,0},{2,2,0},   // block 2
        {0,1,1},{1,1,0},{2,2,0},   // block 3
        {0,1,1},{1,1,0},{2,2,0},   // block 4
    };
    const float* biases[3] = { b0, b1, b2 };
    unsigned short* src = xA;
    unsigned short* dst = xB;
    for (int i = 0; i < 12; ++i) {
        conv_gemm<<<BT / 64, 256, 0, stream>>>(
            src, dst, wtap + (size_t)seq[i].c * 3 * Fn * Fn,
            biases[seq[i].c], seq[i].dil, seq[i].relu);
        unsigned short* t = src; src = dst; dst = t;
    }
    // 12 swaps -> result back in xA (== src)

    linear_gemm<1><<<GEMM_BLOCKS, 256, 0, stream>>>(src, whT, bh, outm);

    fc_wmma<<<(BT / 32) / 8, 256, 0, stream>>>(outm, wfcT, bfc, logits);

    crf_loss<<<Bn, 32, 0, stream>>>(input_ids, labels, logits, crf_start, crf_end, crf_trans,
                                    (float*)d_out);

    (void)in_sizes; (void)n_in; (void)out_size; (void)ws_size;
}
